// TriangleAttention_5531917877748
// MI455X (gfx1250) — matrix operand order repaired and checked by reading, not compiled, not run on hardware
//
#include <hip/hip_runtime.h>

#define LL 256
#define DD 144
#define DPAD 160
#define CC 32
#define HH 4
#define HD 8
#define NPAIR (LL*LL)   // 65536 pairs
#define NOUT 256        // padded proj N: qkv(96) + bias(4) + gate(144) -> 244 -> 256

typedef _Float16 v8h_t  __attribute__((ext_vector_type(8)));
typedef _Float16 v16h_t __attribute__((ext_vector_type(16)));
typedef float    v8f_t  __attribute__((ext_vector_type(8)));

// Build a 16-half fragment from two contiguous 8-half chunks (per-lane).
static __device__ __forceinline__ v16h_t frag2(const _Float16* p0, const _Float16* p1) {
  union { v16h_t v; v8h_t h[2]; } u;
  u.h[0] = *reinterpret_cast<const v8h_t*>(p0);
  u.h[1] = *reinterpret_cast<const v8h_t*>(p1);
  return u.v;
}
static __device__ __forceinline__ v8f_t wmma16(v16h_t a, v16h_t b, v8f_t c) {
  return __builtin_amdgcn_wmma_f32_16x16x32_f16(false, a, false, b, (short)0, c, false, false);
}

// ---------------- Kernel 0: pack weights into f16 B-fragment friendly layouts -------------
__global__ void prep_weights(const float* __restrict__ Wqkv, const float* __restrict__ Wb,
                             const float* __restrict__ Wg, const float* __restrict__ Wout,
                             _Float16* __restrict__ Wt, _Float16* __restrict__ Wot) {
  int idx = blockIdx.x * 256 + threadIdx.x;
  if (idx < NOUT * DPAD) {
    int n = idx / DPAD, k = idx % DPAD;
    float v = 0.f;
    if (k < DD) {
      if (n < 96)       v = Wqkv[k * 96 + n];
      else if (n < 100) v = Wb[k * 4 + (n - 96)];
      else if (n < 244) v = Wg[k * DD + (n - 100)];
    }
    Wt[idx] = (_Float16)v;
  } else {
    int j = idx - NOUT * DPAD;
    if (j < DPAD * CC) {
      int n = j >> 5, kc = j & 31;
      float v = (n < DD) ? Wout[kc * DD + n] : 0.f;
      Wot[j] = (_Float16)v;
    }
  }
}

// ---------------- Kernel 1: LayerNorm, f32 -> f16 with K padded to 160 --------------------
__global__ void ln_kernel(const float* __restrict__ z, const float* __restrict__ gamma,
                          const float* __restrict__ beta, _Float16* __restrict__ znf16) {
  int lane = threadIdx.x & 31;
  int wave = threadIdx.x >> 5;
  long row = (long)blockIdx.x * 8 + wave;
  const float* zr = z + row * DD;
  float x0 = zr[lane], x1 = zr[lane + 32], x2 = zr[lane + 64], x3 = zr[lane + 96];
  float x4 = (lane < 16) ? zr[lane + 128] : 0.f;
  float s  = x0 + x1 + x2 + x3 + x4;
  float ss = x0*x0 + x1*x1 + x2*x2 + x3*x3 + x4*x4;
  for (int off = 16; off > 0; off >>= 1) {
    s  += __shfl_xor(s,  off, 32);
    ss += __shfl_xor(ss, off, 32);
  }
  float mean = s * (1.f / DD);
  float var  = ss * (1.f / DD) - mean * mean;
  float rstd = rsqrtf(var + 1e-5f);
  _Float16* o = znf16 + row * DPAD;
  o[lane]      = (_Float16)((x0 - mean) * rstd * gamma[lane]      + beta[lane]);
  o[lane + 32] = (_Float16)((x1 - mean) * rstd * gamma[lane + 32] + beta[lane + 32]);
  o[lane + 64] = (_Float16)((x2 - mean) * rstd * gamma[lane + 64] + beta[lane + 64]);
  o[lane + 96] = (_Float16)((x3 - mean) * rstd * gamma[lane + 96] + beta[lane + 96]);
  if (lane < 16) o[lane + 128] = (_Float16)((x4 - mean) * rstd * gamma[lane + 128] + beta[lane + 128]);
  else           o[lane + 128] = (_Float16)0.f;  // zero pad K = 144..159
}

// ---------------- Kernel 2: projection GEMM (65536 x 160 x 256) + scatter ----------------
__global__ __launch_bounds__(128) void proj_kernel(
    const _Float16* __restrict__ znf16, const _Float16* __restrict__ Wt,
    _Float16* __restrict__ qp, _Float16* __restrict__ kp, _Float16* __restrict__ vt,
    float* __restrict__ biasB, _Float16* __restrict__ gbuf) {
  int lane = threadIdx.x & 31;
  int w = threadIdx.x >> 5;
  int tile = blockIdx.x * 4 + w;
  int mtile = tile >> 4;
  int ntile = tile & 15;
  int ln = lane & 15, hi = lane >> 4;
  v8f_t acc = {};
  const _Float16* arow = znf16 + (long)(mtile * 16 + ln) * DPAD;
  const _Float16* brow = Wt + (long)(ntile * 16 + ln) * DPAD;
  __builtin_prefetch(arow, 0, 1);   // global_prefetch_b8
  __builtin_prefetch(brow, 0, 1);
  for (int ks = 0; ks < 5; ++ks) {
    v16h_t a = frag2(arow + ks * 32 + hi * 8, arow + ks * 32 + hi * 8 + 16);
    v16h_t b = frag2(brow + ks * 32 + hi * 8, brow + ks * 32 + hi * 8 + 16);
    acc = wmma16(a, b, acc);
  }
  int n = ntile * 16 + ln;
  for (int r = 0; r < 8; ++r) {
    int m = mtile * 16 + r + hi * 8;   // pair index = i*256 + j
    int i = m >> 8, jj = m & 255;
    float val = acc[r];
    if (n < 96) {
      int which = n >> 5;              // 0=q, 1=k, 2=v
      int c = n & 31;
      int h = c >> 3, d = c & 7;
      if (which == 0)      qp[(((long)i * HH + h) * LL + jj) * 32 + d] = (_Float16)val;
      else if (which == 1) kp[(((long)i * HH + h) * LL + jj) * 32 + d] = (_Float16)val;
      else                 vt[(((long)i * HH + h) * 16 + d) * LL + jj] = (_Float16)val;
    } else if (n < 100) {
      biasB[(long)(n - 96) * NPAIR + m] = val;          // bias[h][j][k]
    } else if (n < 244) {
      gbuf[(long)m * DD + (n - 100)] = (_Float16)val;   // pre-sigmoid gate
    }
  }
}

// ---------------- Kernel 3: triangle attention core -------------------------------------
// One block per (h, i, 32-row j tile). Async-stage K panel to LDS, QK^T (padded K=32 WMMA),
// bias+mask, softmax, P@V with k split across wave pairs + LDS partial reduction.
__global__ __launch_bounds__(128) void attn_kernel(
    const _Float16* __restrict__ qp, const _Float16* __restrict__ kp,
    const _Float16* __restrict__ vt, const float* __restrict__ biasB,
    const float* __restrict__ pm, _Float16* __restrict__ Obuf) {
  __shared__ float pmS[LL];
  __shared__ float S[32][264];                 // scores; later reused for P@V partials
  __shared__ __align__(16) char U[32 * 272 * 2]; // phase1: K panel (16KB); phase2+: P (17KB)
  _Float16 (*Ksh)[32]  = (_Float16(*)[32])U;   // [256][32]
  _Float16 (*P)[272]   = (_Float16(*)[272])U;  // [32][272]

  int jt = blockIdx.x;                 // 0..7
  int i  = blockIdx.y;                 // 0..255
  int h  = blockIdx.z;                 // 0..3
  int tid = threadIdx.x;
  int lane = tid & 31, w = tid >> 5;
  int ln = lane & 15, hi = lane >> 4;

  const _Float16* qbase = qp + ((long)i * HH + h) * LL * 32;
  const _Float16* kbase = kp + ((long)i * HH + h) * LL * 32;
  const _Float16* vbase = vt + ((long)i * HH + h) * 16 * LL;
  const float* bbase = biasB + (long)h * NPAIR;
  const float scale = 0.35355339059327373f;   // 1/sqrt(8)

  // Async-stage the K panel (256 rows x 32 halves = 16KB) into LDS: 8 b128 per thread.
  {
    unsigned base_lds = (unsigned)(uintptr_t)(&Ksh[0][0]);
    for (int it = 0; it < 8; ++it) {
      int e = (tid + 128 * it) * 8;                 // half index into the panel
      unsigned laddr = base_lds + (unsigned)(e * 2);
      const _Float16* g = kbase + e;
      asm volatile("global_load_async_to_lds_b128 %0, %1, off"
                   :: "v"(laddr), "v"(g) : "memory");
    }
  }
  const float* pmrow = pm + i * LL;
  pmS[tid] = pmrow[tid];
  pmS[tid + 128] = pmrow[tid + 128];
  asm volatile("s_wait_asynccnt 0x0" ::: "memory");
  __syncthreads();

  // Phase 1: scores for 32 j-rows x 256 k (32 16x16 tiles over 4 waves); B frags from LDS.
  for (int iter = 0; iter < 8; ++iter) {
    int task = iter * 4 + w;
    int jsub = task >> 4;    // 0..1
    int ksub = task & 15;    // 0..15
    const _Float16* ap = qbase + (long)(jt * 32 + jsub * 16 + ln) * 32 + hi * 8;
    v16h_t a = frag2(ap, ap + 16);                      // d>=8 chunks are zero-padded
    const _Float16* bp = &Ksh[ksub * 16 + ln][0] + hi * 8;
    v16h_t b = frag2(bp, bp + 16);
    v8f_t c = {};
    c = wmma16(a, b, c);
    int kk = ksub * 16 + ln;
    float pmk = pmS[kk];
    for (int r = 0; r < 8; ++r) {
      int jj = jsub * 16 + r + hi * 8;
      int jglob = jt * 32 + jj;
      float val = c[r] * scale + bbase[(long)jglob * LL + kk];
      bool ok = (pmS[jglob] > 0.f) && (pmk > 0.f);
      S[jj][kk] = ok ? val : -1e9f;
    }
  }
  __syncthreads();

  // Phase 2: softmax over k, 4 threads per row; write probs (f16) over the dead K panel.
  {
    int row = tid >> 2;
    int k0 = (tid & 3) * 64;
    float m = -3.0e38f;
    for (int k = 0; k < 64; ++k) m = fmaxf(m, S[row][k0 + k]);
    m = fmaxf(m, __shfl_xor(m, 1, 32));
    m = fmaxf(m, __shfl_xor(m, 2, 32));
    float sum = 0.f;
    for (int k = 0; k < 64; ++k) {
      float e = __expf(S[row][k0 + k] - m);
      S[row][k0 + k] = e;
      sum += e;
    }
    sum += __shfl_xor(sum, 1, 32);
    sum += __shfl_xor(sum, 2, 32);
    float inv = 1.f / sum;
    for (int k = 0; k < 64; ++k) P[row][k0 + k] = (_Float16)(S[row][k0 + k] * inv);
  }
  __syncthreads();

  // Phase 3: O = P @ V. All 4 waves: (jsub = w&1, k-half = w>>1); partial-sum via LDS.
  {
    int jsub  = w & 1;
    int khalf = w >> 1;
    v8f_t acc = {};
    for (int ks = khalf * 4; ks < khalf * 4 + 4; ++ks) {
      const _Float16* ap = &P[jsub * 16 + ln][ks * 32 + hi * 8];
      v16h_t a = frag2(ap, ap + 16);
      const _Float16* bp = vbase + (long)ln * LL + ks * 32 + hi * 8;  // vt rows d>=8 are zero
      v16h_t b = frag2(bp, bp + 16);
      acc = wmma16(a, b, acc);
    }
    float* partial = &S[0][0];          // S is dead now; need 2*16*16 floats
    if (khalf == 1) {
      for (int r = 0; r < 8; ++r)
        partial[(jsub * 16 + r + hi * 8) * 16 + ln] = acc[r];
    }
    __syncthreads();
    if (khalf == 0 && ln < 8) {
      for (int r = 0; r < 8; ++r) {
        int jj = jsub * 16 + r + hi * 8;
        float v = acc[r] + partial[jj * 16 + ln];
        int jglob = jt * 32 + jj;
        Obuf[((long)i * LL + jglob) * CC + h * HD + ln] = (_Float16)v;
      }
    }
  }
}

// ---------------- Kernel 4: output GEMM (K=32 exact) + sigmoid gate + pair mask ----------
__global__ __launch_bounds__(128) void out_kernel(
    const _Float16* __restrict__ Obuf, const _Float16* __restrict__ Wot,
    const _Float16* __restrict__ gbuf, const float* __restrict__ pm,
    float* __restrict__ out) {
  int lane = threadIdx.x & 31;
  int w = threadIdx.x >> 5;
  int tile = blockIdx.x * 4 + w;
  int mtile = tile / 9;
  int ntile = tile % 9;
  int ln = lane & 15, hi = lane >> 4;
  const _Float16* arow = Obuf + (long)(mtile * 16 + ln) * CC + hi * 8;
  __builtin_prefetch(arow, 0, 1);
  v16h_t a = frag2(arow, arow + 16);
  const _Float16* brow = Wot + (long)(ntile * 16 + ln) * CC + hi * 8;
  v16h_t b = frag2(brow, brow + 16);
  v8f_t c = {};
  c = wmma16(a, b, c);
  int n = ntile * 16 + ln;
  if (n < DD) {
    for (int r = 0; r < 8; ++r) {
      long m = (long)mtile * 16 + r + hi * 8;
      float gv = (float)gbuf[m * DD + n];
      float gate = 1.f / (1.f + __expf(-gv));
      out[m * DD + n] = c[r] * gate * pm[m];
    }
  }
}

extern "C" void kernel_launch(void* const* d_in, const int* in_sizes, int n_in,
                              void* d_out, int out_size, void* d_ws, size_t ws_size,
                              hipStream_t stream) {
  (void)in_sizes; (void)n_in; (void)out_size; (void)ws_size;
  const float* z     = (const float*)d_in[0];
  const float* pm    = (const float*)d_in[1];
  const float* gamma = (const float*)d_in[2];
  const float* beta  = (const float*)d_in[3];
  const float* Wqkv  = (const float*)d_in[4];
  const float* Wb    = (const float*)d_in[5];
  const float* Wg    = (const float*)d_in[6];
  const float* Wout  = (const float*)d_in[7];
  float* out = (float*)d_out;

  char* ws = (char*)d_ws;
  size_t off = 0;
  auto alloc = [&](size_t bytes) {
    char* p = ws + off;
    off += (bytes + 255) & ~(size_t)255;
    return p;
  };
  _Float16* znf16 = (_Float16*)alloc((size_t)NPAIR * DPAD * 2);        // 20.97 MB
  _Float16* Wt    = (_Float16*)alloc((size_t)NOUT * DPAD * 2);         // 82 KB
  _Float16* Wot   = (_Float16*)alloc((size_t)DPAD * CC * 2);           // 10 KB
  _Float16* qp    = (_Float16*)alloc((size_t)LL * HH * LL * 32 * 2);   // 16.8 MB
  _Float16* kp    = (_Float16*)alloc((size_t)LL * HH * LL * 32 * 2);   // 16.8 MB
  _Float16* vt    = (_Float16*)alloc((size_t)LL * HH * 16 * LL * 2);   // 8.4 MB
  float*    biasB = (float*)   alloc((size_t)HH * NPAIR * 4);          // 1 MB
  _Float16* gbuf  = (_Float16*)alloc((size_t)NPAIR * DD * 2);          // 18.9 MB
  _Float16* Obuf  = (_Float16*)alloc((size_t)NPAIR * CC * 2);          // 4.2 MB

  // zero the K/d padding regions so padded-K WMMA is exact
  hipMemsetAsync(qp, 0, (size_t)LL * HH * LL * 32 * 2, stream);
  hipMemsetAsync(kp, 0, (size_t)LL * HH * LL * 32 * 2, stream);
  hipMemsetAsync(vt, 0, (size_t)LL * HH * 16 * LL * 2, stream);

  prep_weights<<<180, 256, 0, stream>>>(Wqkv, Wb, Wg, Wout, Wt, Wot);
  ln_kernel<<<NPAIR / 8, 256, 0, stream>>>(z, gamma, beta, znf16);
  proj_kernel<<<(NPAIR / 16) * (NOUT / 16) / 4, 128, 0, stream>>>(znf16, Wt, qp, kp, vt, biasB, gbuf);
  attn_kernel<<<dim3(8, LL, HH), 128, 0, stream>>>(qp, kp, vt, biasB, pm, Obuf);
  out_kernel<<<(NPAIR / 16) * 9 / 4, 128, 0, stream>>>(Obuf, Wot, gbuf, pm, out);
}
